// GemLiteLinearTriton_49718541418971
// MI455X (gfx1250) — compile-verified, matrix operand order checked by reading, not run on hardware
//
#include <hip/hip_runtime.h>

typedef _Float16 v16h __attribute__((ext_vector_type(16)));
typedef _Float16 v8h  __attribute__((ext_vector_type(8)));
typedef float    v8f  __attribute__((ext_vector_type(8)));
typedef unsigned v4u  __attribute__((ext_vector_type(4)));
typedef int      v8i  __attribute__((ext_vector_type(8)));
typedef int      v4i  __attribute__((ext_vector_type(4)));

#define K_DIM   8192
#define N_DIM   8192
#define M_DIM   128
#define GROUP   128
#define NGROUPS (K_DIM / GROUP)    // 64
#define KSPLIT  2
#define GPB     (NGROUPS / KSPLIT) // 32 groups per block
#define BM      128
#define BN      128
#define MT      8                  // M-tiles per wave
#define LDSW    136                // 128 + 8 halves pad (produced by TDM pad feature)

__global__ void zero_out_kernel(float* __restrict__ out) {
    const int idx = (blockIdx.x * 256 + threadIdx.x) * 4;
    float4 z = {0.f, 0.f, 0.f, 0.f};
    *(float4*)&out[idx] = z;
}

// x fp32 -> f16 once (reused by all 64 N-blocks; removes convert work from GEMM loop)
__global__ void cvt_x_kernel(const float* __restrict__ x, _Float16* __restrict__ xh) {
    const int idx = (blockIdx.x * 256 + threadIdx.x) * 8;
    const float4 a = *(const float4*)&x[idx];
    const float4 b = *(const float4*)&x[idx + 4];
    v8h h;
    h[0] = (_Float16)a.x; h[1] = (_Float16)a.y; h[2] = (_Float16)a.z; h[3] = (_Float16)a.w;
    h[4] = (_Float16)b.x; h[5] = (_Float16)b.y; h[6] = (_Float16)b.z; h[7] = (_Float16)b.w;
    *(v8h*)&xh[idx] = h;
}

// Issue a TDM load of a 128(rows) x 128(halves) f16 tile into LDS with
// 16B padding per 256B row (-> 136-half LDS row stride).
__device__ __forceinline__ void tdm_load_tile(unsigned lds_off, unsigned long long gaddr) {
    v4u d0;
    d0[0] = 1u;                                     // count=1 (valid), user mode
    d0[1] = lds_off;                                // lds_addr (bytes)
    d0[2] = (unsigned)(gaddr & 0xFFFFFFFFull);      // global_addr[31:0] (tile start)
    d0[3] = (unsigned)((gaddr >> 32) & 0x1FFFFFFull) | (2u << 30); // addr[56:32] | type=2
    v8i d1;
    d1[0] = (1 << 16)     // data_size = 2 bytes
          | (1 << 20)     // pad_enable
          | (5 << 22)     // pad_interval: 64 DWORDs (256B) between pads
          | (3 << 25);    // pad_amount: 4 DWORDs (16B)
    d1[1] = (int)(8192u << 16);  // tensor_dim0[15:0] in bits[63:48]
    d1[2] = (int)(128u  << 16);  // tensor_dim1[15:0] in bits[95:80]
    d1[3] = (int)(128u  << 16);  // tile_dim0 = 128 in bits[127:112]
    d1[4] = 128;                 // tile_dim1 = 128 in bits[143:128]
    d1[5] = 8192;                // tensor_dim0_stride[31:0]
    d1[6] = 0;
    d1[7] = 0;
    v4i z4 = {0, 0, 0, 0};            // groups 2/3 unused (2-D tensor)
    v8i z8 = {0, 0, 0, 0, 0, 0, 0, 0};
    __builtin_amdgcn_tensor_load_to_lds(d0, d1, z4, z4, z8, 0);
}

__global__ __launch_bounds__(256, 2)
void w4a32_gemm_wmma(const _Float16* __restrict__ xh,
                     const int*      __restrict__ Wq,
                     const float*    __restrict__ scales,
                     const float*    __restrict__ zeros,
                     float*          __restrict__ out)
{
    __shared__ __align__(16) _Float16 As[2][BM * LDSW];

    const int tid  = threadIdx.x;
    const int lane = tid & 31;
    const int wave = tid >> 5;

    const int colbase = blockIdx.x * BN;
    const int gbase   = blockIdx.y * GPB;       // K-split group base

    const int nl   = lane & 15;                 // col within 16-wide tile / A row
    const int hi   = lane >> 4;                 // lane half -> K-octet / M+8 on store
    const int ncol = colbase + wave * 16 + nl;  // this lane's output column

    const unsigned lds0 = (unsigned)(uintptr_t)&As[0][0];
    const unsigned lds1 = (unsigned)(uintptr_t)&As[1][0];
    const unsigned long long xbase = (unsigned long long)(uintptr_t)xh;

    v8f acc[MT];
    #pragma unroll
    for (int mt = 0; mt < MT; ++mt) { v8f z = {}; acc[mt] = z; }

    // prologue: DMA first chunk into buffer 0
    if (wave == 0) {
        tdm_load_tile(lds0, xbase + (unsigned long long)(gbase * GROUP) * 2u);
    }

    for (int gi = 0; gi < GPB; ++gi) {
        const int g = gbase + gi;

        if (wave == 0) __builtin_amdgcn_s_wait_tensorcnt(0); // chunk gi landed in LDS
        __syncthreads();                                     // publish to all waves

        // kick DMA for next chunk into the other buffer (it was last read
        // before the barrier above -> no WAR hazard)
        if (wave == 0 && gi + 1 < GPB) {
            tdm_load_tile((gi + 1) & 1 ? lds1 : lds0,
                          xbase + (unsigned long long)((g + 1) * GROUP) * 2u);
        }

        const _Float16* Ab = (gi & 1) ? &As[1][0] : &As[0][0];

        // per-group dequant constants for this lane's column
        const float s  = scales[(size_t)g * N_DIM + ncol];
        const float zs = -zeros[(size_t)g * N_DIM + ncol] * s;

        // packed-word row base: K run starts at g*128 + hi*8 -> word row g*16 + hi
        const int kp0 = g * 16 + hi;

        if (gi + 1 < GPB) {
            __builtin_prefetch(&Wq[(size_t)(kp0 + 16) * N_DIM + ncol], 0, 1);
        }

        #pragma unroll
        for (int step = 0; step < 4; ++step) {
            // B fragment: col = ncol; elems 0..7 <- one packed word (8 consecutive K),
            //             elems 8..15 <- word +2 rows (K+16)
            const unsigned w0 = (unsigned)__builtin_nontemporal_load(
                &Wq[(size_t)(kp0 + step * 4    ) * N_DIM + ncol]);
            const unsigned w1 = (unsigned)__builtin_nontemporal_load(
                &Wq[(size_t)(kp0 + step * 4 + 2) * N_DIM + ncol]);
            v16h b;
            #pragma unroll
            for (int j = 0; j < 8; ++j) {
                b[j]     = (_Float16)__builtin_fmaf((float)((w0 >> (4 * j)) & 15u), s, zs);
                b[j + 8] = (_Float16)__builtin_fmaf((float)((w1 >> (4 * j)) & 15u), s, zs);
            }

            const int kl = step * 32 + hi * 8;  // K offset within LDS chunk
            #pragma unroll
            for (int mt = 0; mt < MT; ++mt) {
                const _Float16* ap = &Ab[(mt * 16 + nl) * LDSW + kl];
                v8h alo = *(const v8h*)ap;          // K = kl .. kl+7
                v8h ahi = *(const v8h*)(ap + 16);   // K = kl+16 .. kl+23
                v16h a = __builtin_shufflevector(alo, ahi,
                         0,1,2,3,4,5,6,7,8,9,10,11,12,13,14,15);
                acc[mt] = __builtin_amdgcn_wmma_f32_16x16x32_f16(
                              false, a, false, b, (short)0, acc[mt], false, false);
            }
        }
    }

    // C/D layout: VGPR v -> M = v (lanes 0-15) or v+8 (lanes 16-31), N = lane%16.
    // Exactly two commutative f32 adds (K-split=2) onto zeroed out -> deterministic.
    #pragma unroll
    for (int mt = 0; mt < MT; ++mt) {
        #pragma unroll
        for (int v = 0; v < 8; ++v) {
            const int row = mt * 16 + v + hi * 8;
            atomicAdd(&out[(size_t)row * N_DIM + ncol], acc[mt][v]);
        }
    }
}

extern "C" void kernel_launch(void* const* d_in, const int* in_sizes, int n_in,
                              void* d_out, int out_size, void* d_ws, size_t ws_size,
                              hipStream_t stream) {
    (void)in_sizes; (void)n_in; (void)ws_size; (void)out_size;
    const float* x      = (const float*)d_in[0];
    const int*   Wq     = (const int*)d_in[1];
    const float* scales = (const float*)d_in[2];
    const float* zeros  = (const float*)d_in[3];
    float*       out    = (float*)d_out;
    _Float16*    xh     = (_Float16*)d_ws;   // 128*8192*2B = 2 MB scratch

    // 1) zero the accumulation target (atomics add onto it)
    zero_out_kernel<<<dim3((M_DIM * N_DIM) / (256 * 4)), dim3(256), 0, stream>>>(out);
    // 2) convert activations to f16 once
    cvt_x_kernel<<<dim3((M_DIM * K_DIM) / (256 * 8)), dim3(256), 0, stream>>>(x, xh);
    // 3) TDM-fed WMMA GEMM
    dim3 grid(N_DIM / BN, KSPLIT);  // (64, 2) = 128 workgroups
    w4a32_gemm_wmma<<<grid, dim3(256), 0, stream>>>(xh, Wq, scales, zeros, out);
}
